// ImprovedHybridReasoningLayer_72103910966039
// MI455X (gfx1250) — matrix operand order repaired and checked by reading, not compiled, not run on hardware
//
#include <hip/hip_runtime.h>
#include <hip/hip_bf16.h>
#include <math.h>

typedef __attribute__((ext_vector_type(16))) __bf16 v16bf;
typedef __attribute__((ext_vector_type(8)))  float  v8f;

constexpr int    Dm  = 512;
constexpr int    Tm  = 128;
constexpr int    Bm  = 512;
constexpr int    HN  = 4;
constexpr int    HDm = 128;
constexpr int    FFm = 2048;
constexpr size_t BT  = (size_t)Bm * Tm;   // 65536 rows
constexpr size_t BTD = BT * Dm;           // 33.5M elems

// ------------------------------------------------------------------
// f32 -> bf16 convert (grid-stride)
// ------------------------------------------------------------------
__global__ void cvt_f32_bf16(const float* __restrict__ s, __bf16* __restrict__ d, size_t n) {
  size_t i  = (size_t)blockIdx.x * blockDim.x + threadIdx.x;
  size_t st = (size_t)gridDim.x * blockDim.x;
  for (; i < n; i += st) d[i] = (__bf16)s[i];
}

// ------------------------------------------------------------------
// Async global->LDS copy (CDNA5): per-lane 16B, tracked on ASYNCcnt.
// LDS byte address = low 32 bits of the generic pointer.
// ------------------------------------------------------------------
__device__ __forceinline__ void async_b128_to_lds(void* lds_ptr, const void* gptr) {
  const unsigned laddr = (unsigned)(size_t)lds_ptr;
  asm volatile("global_load_async_to_lds_b128 %0, %1, off"
               :: "v"(laddr), "v"(gptr) : "memory");
}
__device__ __forceinline__ void wait_asynccnt0() {
  asm volatile("s_wait_asynccnt 0" ::: "memory");
}

// ------------------------------------------------------------------
// Tiled bf16 WMMA GEMM: C[M,N] = A[M,K] * B[K,N] + bias
// block = 256 threads (8 waves), tile 128x64, K-step 32.
// B tile staged TRANSPOSED in LDS so fragments read as 2x ds_load_b128.
// Next K-tile prefetched into registers during compute (overlap).
// MODE 0: f32 out = acc+bias ; MODE 1: bf16 out = gelu(acc+bias)
// ------------------------------------------------------------------
constexpr int SAS = 40;  // LDS stride (bf16) for 128x32 A tile
constexpr int SBT = 40;  // LDS stride (bf16) for 64x32 B^T tile

template <int MODE>
__global__ __launch_bounds__(256) void gemm_bf16(
    const __bf16* __restrict__ A, int lda,
    const __bf16* __restrict__ Bw, int ldb,
    const float* __restrict__ bias,
    float* __restrict__ out32, __bf16* __restrict__ out16, int ldo,
    int K)
{
  __shared__ __bf16 sA[128 * SAS];
  __shared__ __bf16 sBt[64 * SBT];
  const int tid  = threadIdx.x;
  const int lane = tid & 31;
  const int wv   = tid >> 5;
  const int m0   = blockIdx.y * 128;
  const int n0   = blockIdx.x * 64;
  const int rg   = wv >> 1;          // row group 0..3 -> rows 32rg..32rg+31
  const int cg   = wv & 1;           // col group 0..1 -> cols 32cg..32cg+31
  const int l15  = lane & 15;
  const int hf   = lane >> 4;
  const int ar0  = tid >> 2, aseg = tid & 3;  // A: 128 rows x 4 x 16B (2 rows/thread)
  const int brow = tid >> 3, bseg = tid & 7;  // B: 32 rows x 8 x 16B
  v8f acc00 = {}, acc01 = {}, acc10 = {}, acc11 = {};

  const __bf16* Ag = A + (size_t)m0 * lda;
  const __bf16* Bg = Bw + n0;
  // prefetch first K-tile into registers
  uint4 a0 = *(const uint4*)(Ag + (size_t)ar0 * lda + aseg * 8);
  uint4 a1 = *(const uint4*)(Ag + (size_t)(ar0 + 64) * lda + aseg * 8);
  uint4 b0 = *(const uint4*)(Bg + (size_t)brow * ldb + bseg * 8);

  for (int k0 = 0; k0 < K; k0 += 32) {
    // stage current tile (A direct, B transposed)
    *(uint4*)(sA + ar0 * SAS + aseg * 8) = a0;
    *(uint4*)(sA + (ar0 + 64) * SAS + aseg * 8) = a1;
    {
      union { uint4 u; __bf16 e[8]; } tb; tb.u = b0;
#pragma unroll
      for (int j = 0; j < 8; ++j)
        sBt[(bseg * 8 + j) * SBT + brow] = tb.e[j];
    }
    __syncthreads();
    // prefetch next tile while computing
    if (k0 + 32 < K) {
      a0 = *(const uint4*)(Ag + (size_t)ar0 * lda + (k0 + 32) + aseg * 8);
      a1 = *(const uint4*)(Ag + (size_t)(ar0 + 64) * lda + (k0 + 32) + aseg * 8);
      b0 = *(const uint4*)(Bg + (size_t)(k0 + 32 + brow) * ldb + bseg * 8);
    }
    // fragments (both k-contiguous -> 2x b128 each)
    v16bf af0, af1, bf0, bf1;
#pragma unroll
    for (int v = 0; v < 8; ++v) {
      const int kb = hf * 16 + 2 * v;
      const int Mr0 = rg * 32 + l15;
      const int Mr1 = rg * 32 + 16 + l15;
      af0[2 * v] = sA[Mr0 * SAS + kb];     af0[2 * v + 1] = sA[Mr0 * SAS + kb + 1];
      af1[2 * v] = sA[Mr1 * SAS + kb];     af1[2 * v + 1] = sA[Mr1 * SAS + kb + 1];
      const int kk = hf * 16 + 2 * v;
      const int Nc0 = cg * 32 + l15;
      const int Nc1 = cg * 32 + 16 + l15;
      bf0[2 * v] = sBt[Nc0 * SBT + kk];    bf0[2 * v + 1] = sBt[Nc0 * SBT + kk + 1];
      bf1[2 * v] = sBt[Nc1 * SBT + kk];    bf1[2 * v + 1] = sBt[Nc1 * SBT + kk + 1];
    }
    acc00 = __builtin_amdgcn_wmma_f32_16x16x32_bf16(false, af0, false, bf0, (short)0, acc00, false, false);
    acc01 = __builtin_amdgcn_wmma_f32_16x16x32_bf16(false, af0, false, bf1, (short)0, acc01, false, false);
    acc10 = __builtin_amdgcn_wmma_f32_16x16x32_bf16(false, af1, false, bf0, (short)0, acc10, false, false);
    acc11 = __builtin_amdgcn_wmma_f32_16x16x32_bf16(false, af1, false, bf1, (short)0, acc11, false, false);
    __syncthreads();
  }
#pragma unroll
  for (int s = 0; s < 2; ++s)
#pragma unroll
    for (int u = 0; u < 2; ++u) {
      v8f acc = (s == 0) ? (u == 0 ? acc00 : acc01) : (u == 0 ? acc10 : acc11);
      const int Ng = n0 + cg * 32 + u * 16 + l15;
      const float bv = bias ? bias[Ng] : 0.0f;
#pragma unroll
      for (int v = 0; v < 8; ++v) {
        const int Mg = m0 + rg * 32 + s * 16 + v + 8 * hf;
        float x = acc[v] + bv;
        if (MODE == 1) {
          x = 0.5f * x * (1.0f + erff(x * 0.70710678118654752f));  // exact GELU
          out16[(size_t)Mg * ldo + Ng] = (__bf16)x;
        } else {
          out32[(size_t)Mg * ldo + Ng] = x;
        }
      }
    }
}

// ------------------------------------------------------------------
// RoPE + pack to (B,H,T,HD) bf16 head layout
// ------------------------------------------------------------------
__global__ void rope_pack(const float* __restrict__ Qf, const float* __restrict__ Kf,
                          const float* __restrict__ Vf,
                          __bf16* __restrict__ Qb, __bf16* __restrict__ Kb,
                          __bf16* __restrict__ Vb)
{
  size_t idx = (size_t)blockIdx.x * 256 + threadIdx.x;
  const int i = (int)(idx & 63);  idx >>= 6;   // rotation pair index (HD/2)
  const int h = (int)(idx & 3);   idx >>= 2;
  const int t = (int)(idx & 127); idx >>= 7;
  const int b = (int)idx;
  const float invf = expf(-(float)(2 * i) * (9.210340371976184f / 128.0f)); // 10000^(-2i/HD)
  float sn, cs;
  sincosf((float)t * invf, &sn, &cs);
  const size_t src = ((size_t)b * Tm + t) * Dm + h * HDm;
  const size_t dst = (((size_t)b * HN + h) * Tm + t) * HDm;
  const float q1 = Qf[src + i], q2 = Qf[src + i + 64];
  const float k1 = Kf[src + i], k2 = Kf[src + i + 64];
  Qb[dst + i]      = (__bf16)(q1 * cs - q2 * sn);
  Qb[dst + i + 64] = (__bf16)(q2 * cs + q1 * sn);
  Kb[dst + i]      = (__bf16)(k1 * cs - k2 * sn);
  Kb[dst + i + 64] = (__bf16)(k2 * cs + k1 * sn);
  Vb[dst + i]      = (__bf16)Vf[src + i];
  Vb[dst + i + 64] = (__bf16)Vf[src + i + 64];
}

// ------------------------------------------------------------------
// Causal attention, one (b,h) per block, 8 waves, all tiles in LDS.
// Q/K tiles copied via async global->LDS (ASYNCcnt); V staged
// transposed (VGPR roundtrip) so P@V B-fragments are k-contiguous.
// scores = QK^T/sqrt(HD) -> causal softmax -> P@V, WMMA both GEMMs.
// ------------------------------------------------------------------
constexpr int SQS = 136;  // bf16 LDS stride for 128x128 tiles
constexpr int SSS = 132;  // f32 LDS stride for scores
constexpr size_t SMEM_ATTN = (size_t)3 * 128 * SQS * 2 + (size_t)128 * SSS * 4;

__global__ __launch_bounds__(256) void attn_kernel(
    const __bf16* __restrict__ Qb, const __bf16* __restrict__ Kb,
    const __bf16* __restrict__ Vb, float* __restrict__ out)
{
  extern __shared__ char smem[];
  __bf16* sQ  = (__bf16*)smem;
  __bf16* sK  = sQ + 128 * SQS;
  __bf16* sVt = sK + 128 * SQS;          // transposed: sVt[d][s]
  float*  sS  = (float*)(sVt + 128 * SQS);
  __bf16* sP  = sQ;   // probs overlay Q (Q dead after scores)

  const int bh = blockIdx.x;
  const int b  = bh >> 2;
  const int h  = bh & 3;
  const size_t base = (size_t)bh * Tm * HDm;
  const int tid = threadIdx.x;
  // 128x128 bf16 tile = 2048 uint4; row = 16 uint4
  for (int i = tid; i < 2048; i += 256) {
    const int row = i >> 4, seg = i & 15;
    // Q/K: straight copies -> async global->LDS b128 (no VGPR roundtrip)
    async_b128_to_lds(sQ + row * SQS + seg * 8, Qb + base + row * HDm + seg * 8);
    async_b128_to_lds(sK + row * SQS + seg * 8, Kb + base + row * HDm + seg * 8);
    // V: transpose during staging (per-element scatter)
    union { uint4 u; __bf16 e[8]; } tv;
    tv.u = *(const uint4*)(Vb + base + row * HDm + seg * 8);
#pragma unroll
    for (int j = 0; j < 8; ++j)
      sVt[(seg * 8 + j) * SQS + row] = tv.e[j];   // transpose: d-major
  }
  wait_asynccnt0();
  __syncthreads();
  const int lane = tid & 31, wv = tid >> 5;
  const int l15 = lane & 15, hf = lane >> 4;
  const int Mr = wv * 16 + l15;

  // ---- scores: A=Q, B=K^T (B[k][n] = K[n][k], contiguous k gathers) ----
  {
    v8f acc[8] = {};
    for (int k0 = 0; k0 < HDm; k0 += 32) {
      v16bf af;
#pragma unroll
      for (int v = 0; v < 8; ++v) {
        const int kb = k0 + hf * 16 + 2 * v;
        af[2 * v]     = sQ[Mr * SQS + kb];
        af[2 * v + 1] = sQ[Mr * SQS + kb + 1];
      }
#pragma unroll
      for (int j = 0; j < 8; ++j) {
        v16bf bfr;
        const int Nc = j * 16 + l15;
#pragma unroll
        for (int v = 0; v < 8; ++v) {
          const int kk = k0 + hf * 16 + 2 * v;
          bfr[2 * v]     = sK[Nc * SQS + kk];
          bfr[2 * v + 1] = sK[Nc * SQS + kk + 1];
        }
        acc[j] = __builtin_amdgcn_wmma_f32_16x16x32_bf16(false, af, false, bfr, (short)0, acc[j], false, false);
      }
    }
#pragma unroll
    for (int j = 0; j < 8; ++j)
#pragma unroll
      for (int v = 0; v < 8; ++v)
        sS[(wv * 16 + v + 8 * hf) * SSS + j * 16 + l15] = acc[j][v];
  }
  __syncthreads();

  // ---- causal softmax, one row per thread (f32 in LDS) ----
  if (tid < 128) {
    const int t = tid;
    const float scale = 0.08838834764831845f;  // 1/sqrt(128)
    float m = -3.0e38f;
    for (int s = 0; s <= t; ++s) m = fmaxf(m, sS[t * SSS + s] * scale);
    float sum = 0.0f;
    for (int s = 0; s <= t; ++s) sum += expf(sS[t * SSS + s] * scale - m);
    const float inv = 1.0f / sum;
    for (int s = 0; s < 128; ++s)
      sP[t * SQS + s] = (s <= t) ? (__bf16)(expf(sS[t * SSS + s] * scale - m) * inv)
                                 : (__bf16)0.0f;
  }
  __syncthreads();

  // ---- P @ V (B-frags from transposed V: contiguous k) ----
  v8f acc[8] = {};
  for (int k0 = 0; k0 < Tm; k0 += 32) {
    v16bf af;
#pragma unroll
    for (int v = 0; v < 8; ++v) {
      const int kb = k0 + hf * 16 + 2 * v;
      af[2 * v]     = sP[Mr * SQS + kb];
      af[2 * v + 1] = sP[Mr * SQS + kb + 1];
    }
#pragma unroll
    for (int j = 0; j < 8; ++j) {
      v16bf bfr;
      const int Nc = j * 16 + l15;
#pragma unroll
      for (int v = 0; v < 8; ++v) {
        const int kk = k0 + hf * 16 + 2 * v;
        bfr[2 * v]     = sVt[Nc * SQS + kk];
        bfr[2 * v + 1] = sVt[Nc * SQS + kk + 1];
      }
      acc[j] = __builtin_amdgcn_wmma_f32_16x16x32_bf16(false, af, false, bfr, (short)0, acc[j], false, false);
    }
  }
#pragma unroll
  for (int j = 0; j < 8; ++j)
#pragma unroll
    for (int v = 0; v < 8; ++v) {
      const int t = wv * 16 + v + 8 * hf;
      const int d = j * 16 + l15;
      out[((size_t)b * Tm + t) * Dm + h * HDm + d] = acc[j][v];  // (B,T,D) merge-heads
    }
}

// ------------------------------------------------------------------
// LayerNorm over D=512: y = LN(a + alpha*b)*g + bb ; optional bf16 copy
// ------------------------------------------------------------------
__global__ __launch_bounds__(256) void ln512(
    const float* __restrict__ a, long long sa,
    const float* __restrict__ b, long long sb, float alpha,
    const float* __restrict__ g, const float* __restrict__ bb,
    float* __restrict__ y32, long long sy32,
    __bf16* __restrict__ y16, long long sy16)
{
  __shared__ float red[256];
  const int row = blockIdx.x;
  const int tid = threadIdx.x;
  const float* ar = a + (size_t)row * sa;
  float v0 = ar[tid], v1 = ar[tid + 256];
  if (b) {
    const float* br = b + (size_t)row * sb;
    v0 += alpha * br[tid]; v1 += alpha * br[tid + 256];
  }
  red[tid] = v0 + v1;
  __syncthreads();
  for (int st = 128; st > 0; st >>= 1) { if (tid < st) red[tid] += red[tid + st]; __syncthreads(); }
  const float mu = red[0] * (1.0f / 512.0f);
  __syncthreads();
  const float d0 = v0 - mu, d1 = v1 - mu;
  red[tid] = d0 * d0 + d1 * d1;
  __syncthreads();
  for (int st = 128; st > 0; st >>= 1) { if (tid < st) red[tid] += red[tid + st]; __syncthreads(); }
  const float inv = rsqrtf(red[0] * (1.0f / 512.0f) + 1e-5f);
  const float o0 = d0 * inv * g[tid] + bb[tid];
  const float o1 = d1 * inv * g[tid + 256] + bb[tid + 256];
  if (y32) { y32[(size_t)row * sy32 + tid] = o0; y32[(size_t)row * sy32 + tid + 256] = o1; }
  if (y16) { y16[(size_t)row * sy16 + tid] = (__bf16)o0; y16[(size_t)row * sy16 + tid + 256] = (__bf16)o1; }
}

// ------------------------------------------------------------------
// depthwise causal convs (local K=4 direct, EMA via exact recurrence
// since T=128 <= EMA_K) + pack combined (B,T,3D) bf16
// ------------------------------------------------------------------
__global__ void conv_kernel(const float* __restrict__ x1, const float* __restrict__ lw,
                            const float* __restrict__ lb, __bf16* __restrict__ comb)
{
  const int idx = blockIdx.x * 256 + threadIdx.x;  // b*D + d
  const int d = idx & 511;
  const int b = idx >> 9;
  const float w0 = lw[d * 4 + 0], w1 = lw[d * 4 + 1], w2 = lw[d * 4 + 2], w3 = lw[d * 4 + 3];
  const float lbias = lb[d];
  float xm1 = 0.f, xm2 = 0.f, xm3 = 0.f, ema = 0.f;
  const float* xr = x1 + (size_t)b * Tm * Dm + d;
  __bf16* cr = comb + (size_t)b * Tm * (3 * Dm) + d;
  for (int t = 0; t < Tm; ++t) {
    const float xt  = xr[(size_t)t * Dm];
    const float loc = w3 * xt + w2 * xm1 + w1 * xm2 + w0 * xm3 + lbias;
    ema = 0.9f * ema + 0.1f * xt;  // (1-d)*d^lag taps, full history within T
    cr[(size_t)t * 1536]        = (__bf16)xt;
    cr[(size_t)t * 1536 + 512]  = (__bf16)loc;
    cr[(size_t)t * 1536 + 1024] = (__bf16)ema;
    xm3 = xm2; xm2 = xm1; xm1 = xt;
  }
}

// deltas_raw = sigmoid(gate_act) * tanh(delta_act)  (in-place on da)
__global__ void prod_kernel(const float* __restrict__ ga, float* __restrict__ da, size_t n) {
  size_t i  = (size_t)blockIdx.x * blockDim.x + threadIdx.x;
  size_t st = (size_t)gridDim.x * blockDim.x;
  for (; i < n; i += st) {
    const float s = 1.0f / (1.0f + expf(-ga[i]));
    da[i] = s * tanhf(da[i]);
  }
}

// pre_states = seq_state + cumsum_t(deltas) * STEP
__global__ void cumsum_kernel(const float* __restrict__ deltas, const float* __restrict__ st0,
                              float* __restrict__ pre)
{
  const int idx = blockIdx.x * 256 + threadIdx.x;
  const int d = idx & 511;
  const int b = idx >> 9;
  float acc = 0.0f;
  const float s0 = st0[(size_t)b * Dm + d];
  const float* dr = deltas + (size_t)b * Tm * Dm + d;
  float* pr = pre + (size_t)b * Tm * Dm + d;
  for (int t = 0; t < Tm; ++t) {
    acc += dr[(size_t)t * Dm];
    pr[(size_t)t * Dm] = s0 + acc * 0.25f;  // STEP = 0.5/sqrt(128/32)
  }
}

// out_bf16 = a + alpha*b
__global__ void axpy_pack(const float* __restrict__ a, const float* __restrict__ b, float alpha,
                          __bf16* __restrict__ o, size_t n) {
  size_t i  = (size_t)blockIdx.x * blockDim.x + threadIdx.x;
  size_t st = (size_t)gridDim.x * blockDim.x;
  for (; i < n; i += st) o[i] = (__bf16)(a[i] + alpha * b[i]);
}

// ------------------------------------------------------------------
// Host orchestration
// ------------------------------------------------------------------
extern "C" void kernel_launch(void* const* d_in, const int* in_sizes, int n_in,
                              void* d_out, int out_size, void* d_ws, size_t ws_size,
                              hipStream_t stream)
{
  (void)in_sizes; (void)n_in; (void)out_size; (void)ws_size;
  const float* x_in    = (const float*)d_in[0];
  const float* seqst   = (const float*)d_in[1];
  const float* q_w     = (const float*)d_in[2];  const float* q_b     = (const float*)d_in[3];
  const float* k_w     = (const float*)d_in[4];  const float* k_b     = (const float*)d_in[5];
  const float* v_w     = (const float*)d_in[6];  const float* v_b     = (const float*)d_in[7];
  const float* o_w     = (const float*)d_in[8];  const float* o_b     = (const float*)d_in[9];
  const float* local_w = (const float*)d_in[10]; const float* local_b = (const float*)d_in[11];
  const float* delta_w = (const float*)d_in[12]; const float* delta_b = (const float*)d_in[13];
  const float* gate_w  = (const float*)d_in[14]; const float* gate_b  = (const float*)d_in[15];
  const float* dn_g    = (const float*)d_in[16]; const float* dn_b    = (const float*)d_in[17];
  const float* ln1_g   = (const float*)d_in[18]; const float* ln1_b   = (const float*)d_in[19];
  const float* ln2_g   = (const float*)d_in[20]; const float* ln2_b   = (const float*)d_in[21];
  const float* ln3_g   = (const float*)d_in[22]; const float* ln3_b   = (const float*)d_in[23];
  const float* ln4_g   = (const float*)d_in[24]; const float* ln4_b   = (const float*)d_in[25];
  const float* seq_w   = (const float*)d_in[26]; const float* seq_b   = (const float*)d_in[27];
  const float* ffn_w1  = (const float*)d_in[28]; const float* ffn_b1  = (const float*)d_in[29];
  const float* ffn_w2  = (const float*)d_in[30]; const float* ffn_b2  = (const float*)d_in[31];
  // d_in[32] = ema_w: taps are exactly (1-DECAY)*DECAY^lag -> recurrence in conv_kernel

  char* ws = (char*)d_ws;
  size_t off = 0;
  auto take = [&](size_t bytes) { size_t r = off; off = (off + bytes + 255) & ~(size_t)255; return r; };

  // bf16 weight slabs
  const size_t wqo = take((size_t)Dm * Dm * 2);
  const size_t wko = take((size_t)Dm * Dm * 2);
  const size_t wvo = take((size_t)Dm * Dm * 2);
  const size_t woo = take((size_t)Dm * Dm * 2);
  const size_t wdo = take((size_t)3 * Dm * Dm * 2);
  const size_t wgo = take((size_t)3 * Dm * Dm * 2);
  const size_t wso = take((size_t)Dm * Dm * 2);
  const size_t w1o = take((size_t)Dm * FFm * 2);
  const size_t w2o = take((size_t)FFm * Dm * 2);
  // activation slabs (reused across phases)
  const size_t XB = take(BTD * 2);             // bf16 x / bf16 attn_out
  const size_t F0 = take(BTD * 4);             // Qf32 / attn_out / gate_act / pre-states / states
  const size_t F1 = take(BTD * 4);             // Kf32 / x1
  const size_t F2 = take(BTD * 4);             // Vf32 / o_out / delta_act / deltas / seq / ffn_out
  const size_t G0 = take(BTD * 2);             // Qb (B,H,T,HD)
  const size_t G1 = take(BTD * 2);             // Kb / states bf16
  const size_t G2 = take(BTD * 2);             // Vb / ffn_in bf16
  const size_t HH = take(BT * FFm * 2);        // combined (B,T,3D) then h (B,T,FF)

  auto bfp = [&](size_t o) { return (__bf16*)(ws + o); };
  auto fp  = [&](size_t o) { return (float*)(ws + o); };

  // ---- convert weights + x to bf16 ----
  cvt_f32_bf16<<<1024, 256, 0, stream>>>(q_w,    bfp(wqo), (size_t)Dm * Dm);
  cvt_f32_bf16<<<1024, 256, 0, stream>>>(k_w,    bfp(wko), (size_t)Dm * Dm);
  cvt_f32_bf16<<<1024, 256, 0, stream>>>(v_w,    bfp(wvo), (size_t)Dm * Dm);
  cvt_f32_bf16<<<1024, 256, 0, stream>>>(o_w,    bfp(woo), (size_t)Dm * Dm);
  cvt_f32_bf16<<<2048, 256, 0, stream>>>(delta_w, bfp(wdo), (size_t)3 * Dm * Dm);
  cvt_f32_bf16<<<2048, 256, 0, stream>>>(gate_w,  bfp(wgo), (size_t)3 * Dm * Dm);
  cvt_f32_bf16<<<1024, 256, 0, stream>>>(seq_w,  bfp(wso), (size_t)Dm * Dm);
  cvt_f32_bf16<<<4096, 256, 0, stream>>>(ffn_w1, bfp(w1o), (size_t)Dm * FFm);
  cvt_f32_bf16<<<4096, 256, 0, stream>>>(ffn_w2, bfp(w2o), (size_t)FFm * Dm);
  cvt_f32_bf16<<<8192, 256, 0, stream>>>(x_in,   bfp(XB), BTD);

  const dim3 g512(Dm / 64, (unsigned)(BT / 128));     // (8,512), tile 128x64
  const dim3 g2048(FFm / 64, (unsigned)(BT / 128));   // (32,512)

  // ---- QKV projections ----
  gemm_bf16<0><<<g512, 256, 0, stream>>>(bfp(XB), Dm, bfp(wqo), Dm, q_b, fp(F0), nullptr, Dm, Dm);
  gemm_bf16<0><<<g512, 256, 0, stream>>>(bfp(XB), Dm, bfp(wko), Dm, k_b, fp(F1), nullptr, Dm, Dm);
  gemm_bf16<0><<<g512, 256, 0, stream>>>(bfp(XB), Dm, bfp(wvo), Dm, v_b, fp(F2), nullptr, Dm, Dm);

  // ---- RoPE + head pack ----
  rope_pack<<<(unsigned)((size_t)Bm * Tm * HN * 64 / 256), 256, 0, stream>>>(
      fp(F0), fp(F1), fp(F2), bfp(G0), bfp(G1), bfp(G2));

  // ---- attention ----
  hipFuncSetAttribute(reinterpret_cast<const void*>(&attn_kernel),
                      hipFuncAttributeMaxDynamicSharedMemorySize, (int)SMEM_ATTN);
  attn_kernel<<<Bm * HN, 256, SMEM_ATTN, stream>>>(bfp(G0), bfp(G1), bfp(G2), fp(F0));

  // ---- O projection + LN1 residual ----
  cvt_f32_bf16<<<8192, 256, 0, stream>>>(fp(F0), bfp(XB), BTD);
  gemm_bf16<0><<<g512, 256, 0, stream>>>(bfp(XB), Dm, bfp(woo), Dm, o_b, fp(F2), nullptr, Dm, Dm);
  ln512<<<(unsigned)BT, 256, 0, stream>>>(x_in, Dm, fp(F2), Dm, 1.0f, ln1_g, ln1_b,
                                          fp(F1), Dm, nullptr, 0);           // x1 -> F1

  // ---- depthwise convs -> combined bf16 (B,T,3D) in HH ----
  conv_kernel<<<(Bm * Dm) / 256, 256, 0, stream>>>(fp(F1), local_w, local_b, bfp(HH));

  // ---- gate/delta GEMMs (K=1536) ----
  gemm_bf16<0><<<g512, 256, 0, stream>>>(bfp(HH), 3 * Dm, bfp(wgo), Dm, gate_b, fp(F0), nullptr, Dm, 3 * Dm);
  gemm_bf16<0><<<g512, 256, 0, stream>>>(bfp(HH), 3 * Dm, bfp(wdo), Dm, delta_b, fp(F2), nullptr, Dm, 3 * Dm);
  prod_kernel<<<8192, 256, 0, stream>>>(fp(F0), fp(F2), BTD);                // deltas_raw -> F2
  ln512<<<(unsigned)BT, 256, 0, stream>>>(fp(F2), Dm, nullptr, 0, 0.f, dn_g, dn_b,
                                          fp(F2), Dm, nullptr, 0);           // deltas

  // ---- cumsum + states (LN3) + final_state (LN4) ----
  cumsum_kernel<<<(Bm * Dm) / 256, 256, 0, stream>>>(fp(F2), seqst, fp(F0)); // pre-states -> F0
  ln512<<<(unsigned)BT, 256, 0, stream>>>(fp(F0), Dm, nullptr, 0, 0.f, ln3_g, ln3_b,
                                          fp(F0), Dm, bfp(G1), Dm);          // states f32 + bf16
  ln512<<<Bm, 256, 0, stream>>>(fp(F0) + (size_t)(Tm - 1) * Dm, (long long)Tm * Dm,
                                nullptr, 0, 0.f, ln4_g, ln4_b,
                                (float*)d_out + BTD, Dm, nullptr, 0);        // final_state

  // ---- seq guidance + FFN ----
  gemm_bf16<0><<<g512, 256, 0, stream>>>(bfp(G1), Dm, bfp(wso), Dm, seq_b, fp(F2), nullptr, Dm, Dm);
  axpy_pack<<<8192, 256, 0, stream>>>(fp(F1), fp(F2), 0.3f, bfp(G2), BTD);   // ffn_in bf16
  gemm_bf16<1><<<g2048, 256, 0, stream>>>(bfp(G2), Dm, bfp(w1o), FFm, ffn_b1,
                                          nullptr, bfp(HH), FFm, Dm);        // h = gelu(...)
  gemm_bf16<0><<<g512, 256, 0, stream>>>(bfp(HH), FFm, bfp(w2o), Dm, ffn_b2,
                                         fp(F2), nullptr, Dm, FFm);          // ffn_out
  ln512<<<(unsigned)BT, 256, 0, stream>>>(fp(F1), Dm, fp(F2), Dm, 1.0f, ln2_g, ln2_b,
                                          (float*)d_out, Dm, nullptr, 0);    // out
}
